// RelationExtractor_88373247082924
// MI455X (gfx1250) — matrix operand order repaired and checked by reading, not compiled, not run on hardware
//
#include <hip/hip_runtime.h>
#include <math.h>

// Problem constants from the reference
#define BATCH 64
#define SEQ   512
#define DIM   768
#define NEGV  -1e30f

typedef __attribute__((ext_vector_type(16))) _Float16 v16h;
typedef __attribute__((ext_vector_type(8)))  float    v8f;

__device__ __forceinline__ float sigmoidf_(float x) { return 1.0f / (1.0f + expf(-x)); }

// ---------------------------------------------------------------------------
// One-time weight prep: Wt(N x K) f16 = transpose of W(K x N) f32.
// Makes each WMMA B-fragment (16 consecutive K for one column) a single
// contiguous 32-byte load in the GEMM inner loop.
// ---------------------------------------------------------------------------
__global__ void convert_transpose_kernel(const float* __restrict__ W,
                                         _Float16* __restrict__ Wt,
                                         int K, int N) {
  int i = blockIdx.x * blockDim.x + threadIdx.x;   // over N*K
  int n = i / K, k = i % K;
  Wt[i] = (_Float16)W[(size_t)k * N + n];
}

// ---------------------------------------------------------------------------
// Masked max-pool over sequence window [start, end] (inclusive).
// One block per (b, k); threads stride over D (coalesced). Also scatters
// entity k into its per-entity state buffer e_k (B x D).
// ---------------------------------------------------------------------------
__global__ void pool_kernel(const float* __restrict__ enc, const int* __restrict__ ent_pos,
                            float* __restrict__ ent,
                            float* __restrict__ e0, float* __restrict__ e1, float* __restrict__ e2) {
  int bk = blockIdx.x;            // 0 .. B*3-1
  int b = bk / 3, k = bk % 3;
  int start = ent_pos[bk * 2 + 0];
  int end   = ent_pos[bk * 2 + 1];
  const float* base = enc + (size_t)b * SEQ * DIM;
  float* edst = (k == 0) ? e0 : (k == 1) ? e1 : e2;
  for (int d = threadIdx.x; d < DIM; d += blockDim.x) {
    float m = NEGV;
    for (int s = start; s <= end; ++s)
      m = fmaxf(m, base[(size_t)s * DIM + d]);
    ent[(size_t)bk * DIM + d] = m;
    edst[(size_t)b * DIM + d] = m;
  }
}

// ---------------------------------------------------------------------------
// WMMA GEMM: out(M x N) = epilogue( [A0|A1](M x (K0+K1)) @ W((K0+K1) x N) + bias )
// W supplied pre-transposed + converted: Wt(N x Ktot) f16, so the B fragment
// per lane is one contiguous v16h load.
//   ACT: 1 = relu, 2 = sigmoid, 3 = gate blend  out = g*E + (1-g)*U
// One wave per 16x16 tile, 4 waves/block, EXEC always full (ISA 7.12 req).
// M, N multiples of 16; K0, K1 multiples of 32 (K-blocks never straddle the
// concat boundary). A/B use contiguous k order: lane half h holds k 16h..16h+15.
// ---------------------------------------------------------------------------
template <int ACT>
__global__ void wmma_gemm(const float* __restrict__ A0, int lda0, int K0,
                          const float* __restrict__ A1, int lda1, int K1,
                          const _Float16* __restrict__ Wt,
                          const float* __restrict__ bias,
                          const float* __restrict__ E, const float* __restrict__ U,
                          float* __restrict__ out, int ldo, int N) {
  const int lane = threadIdx.x & 31;
  const int wave = threadIdx.x >> 5;
  const int tilesN = N >> 4;
  const int tile = blockIdx.x * 4 + wave;
  const int mi = tile / tilesN;
  const int ni = tile % tilesN;
  const bool hi  = lane >= 16;
  const int  l15 = lane & 15;
  const int  mrow = mi * 16 + l15;   // A row held by this lane
  const int  ncol = ni * 16 + l15;   // B/C/D column held by this lane
  const int  Ktot = K0 + K1;
  const _Float16* wcol = Wt + (size_t)ncol * Ktot;  // 32B-aligned column

  v8f acc = {};
  for (int k0 = 0; k0 < Ktot; k0 += 32) {
    const float* arow = (k0 < K0) ? (A0 + (size_t)mrow * lda0 + k0)
                                  : (A1 + (size_t)mrow * lda1 + (k0 - K0));
    // A 16x32 f16: lanes 0-15 -> K 0..15; lanes 16-31 -> K 16..31
    const int o0 = hi ? 16 : 0;
    const int o1 = hi ? 24 : 8;
    const float4* p0 = reinterpret_cast<const float4*>(arow + o0);
    const float4* p1 = reinterpret_cast<const float4*>(arow + o1);
    float4 x0 = p0[0], x1 = p0[1], y0 = p1[0], y1 = p1[1];
    v16h a;
    a[0]=(_Float16)x0.x;  a[1]=(_Float16)x0.y;  a[2]=(_Float16)x0.z;  a[3]=(_Float16)x0.w;
    a[4]=(_Float16)x1.x;  a[5]=(_Float16)x1.y;  a[6]=(_Float16)x1.z;  a[7]=(_Float16)x1.w;
    a[8]=(_Float16)y0.x;  a[9]=(_Float16)y0.y;  a[10]=(_Float16)y0.z; a[11]=(_Float16)y0.w;
    a[12]=(_Float16)y1.x; a[13]=(_Float16)y1.y; a[14]=(_Float16)y1.z; a[15]=(_Float16)y1.w;
    // B 32x16 f16: lane holds column ncol; lanes 0-15 K 0..15, lanes 16-31 K 16..31
    const int kb = k0 + (hi ? 16 : 0);
    v16h bv = *reinterpret_cast<const v16h*>(wcol + kb);  // 2x global_load_b128
    acc = __builtin_amdgcn_wmma_f32_16x16x32_f16(false, a, false, bv,
                                                 (short)0, acc, false, false);
  }

  const float bcol = bias[ncol];
#pragma unroll
  for (int r = 0; r < 8; ++r) {
    const int row = mi * 16 + r + (hi ? 8 : 0);
    float x = acc[r] + bcol;
    if (ACT == 1) x = fmaxf(x, 0.0f);
    if (ACT == 2) x = sigmoidf_(x);
    if (ACT == 3) {
      const float g  = sigmoidf_(x);
      const float ev = E[(size_t)row * ldo + ncol];
      const float uv = U[(size_t)row * ldo + ncol];
      x = g * ev + (1.0f - g) * uv;
    }
    out[(size_t)row * ldo + ncol] = x;
  }
}

// ---------------------------------------------------------------------------
// Tiny N=5 GEMM (scalar; a WMMA tile would be 11/16 idle): one thread per
// (m, n). act: 0 = none, 2 = sigmoid.
// ---------------------------------------------------------------------------
__global__ void small_gemm(const float* __restrict__ Hm, const float* __restrict__ W,
                           const float* __restrict__ bias, float* __restrict__ out,
                           int M, int K, int N, int act) {
  int i = blockIdx.x * blockDim.x + threadIdx.x;
  if (i >= M * N) return;
  int m = i / N, n = i % N;
  float acc = bias[n];
  for (int k = 0; k < K; ++k) acc += Hm[(size_t)m * K + k] * W[(size_t)k * N + n];
  if (act == 2) acc = sigmoidf_(acc);
  out[i] = acc;
}

// u = (v @ Ar_W + Ar_b) * e    v:(B,5)  Ar_W:(5,D)  e,u:(B,D)
__global__ void u_kernel(const float* __restrict__ v, const float* __restrict__ ArW,
                         const float* __restrict__ Arb, const float* __restrict__ e,
                         float* __restrict__ u) {
  int b = blockIdx.x;
  for (int d = threadIdx.x; d < DIM; d += blockDim.x) {
    float s = Arb[d];
#pragma unroll
    for (int j = 0; j < 5; ++j) s += v[b * 5 + j] * ArW[j * DIM + d];
    u[(size_t)b * DIM + d] = s * e[(size_t)b * DIM + d];
  }
}

// ent_score = sigmoid(ent @ proj_W + proj_b); one wave32 per (b,k), shuffle reduce.
__global__ void score_kernel(const float* __restrict__ ent, const float* __restrict__ projW,
                             const float* __restrict__ projb, float* __restrict__ out) {
  int bk = blockIdx.x;
  int lane = threadIdx.x;
  float s = 0.0f;
  for (int d = lane; d < DIM; d += 32) s += ent[(size_t)bk * DIM + d] * projW[d];
#pragma unroll
  for (int off = 16; off > 0; off >>= 1) s += __shfl_xor(s, off, 32);
  if (lane == 0) out[bk] = sigmoidf_(s + projb[0]);
}

// final_ent[b, i, d] = e_i[b, d]
__global__ void pack_final(const float* __restrict__ e0, const float* __restrict__ e1,
                           const float* __restrict__ e2, float* __restrict__ out) {
  int idx = blockIdx.x * blockDim.x + threadIdx.x;   // < B*3*D
  int d = idx % DIM;
  int bi = idx / DIM;
  int i = bi % 3, b = bi / 3;
  const float* e = (i == 0) ? e0 : (i == 1) ? e1 : e2;
  out[idx] = e[(size_t)b * DIM + d];
}

// ---------------------------------------------------------------------------
extern "C" void kernel_launch(void* const* d_in, const int* in_sizes, int n_in,
                              void* d_out, int out_size, void* d_ws, size_t ws_size,
                              hipStream_t stream) {
  const float* enc    = (const float*)d_in[0];
  const int*   entpos = (const int*)  d_in[1];
  const float* ArW    = (const float*)d_in[2];
  const float* Arb    = (const float*)d_in[3];
  const float* VrW1   = (const float*)d_in[4];
  const float* Vrb1   = (const float*)d_in[5];
  const float* VrW2   = (const float*)d_in[6];
  const float* Vrb2   = (const float*)d_in[7];
  const float* gateW  = (const float*)d_in[8];
  const float* gateb  = (const float*)d_in[9];
  const float* predW1 = (const float*)d_in[10];
  const float* predb1 = (const float*)d_in[11];
  const float* predW2 = (const float*)d_in[12];
  const float* predb2 = (const float*)d_in[13];
  const float* projW  = (const float*)d_in[14];
  const float* projb  = (const float*)d_in[15];
  float* out = (float*)d_out;   // [rel B*5 | ent_score B*3 | final_ent B*3*D]

  // Workspace carve-up (floats). Total ~9.2 MB.
  float* ws = (float*)d_ws;
  size_t off = 0;
  float* ent = ws + off;                 off += (size_t)BATCH * 3 * DIM;
  float* ebuf[6];
  for (int i = 0; i < 6; ++i) { ebuf[i] = ws + off; off += (size_t)BATCH * DIM; }
  float* ubuf[3];
  for (int i = 0; i < 3; ++i) { ubuf[i] = ws + off; off += (size_t)BATCH * DIM; }
  float* hbuf[3];
  for (int i = 0; i < 3; ++i) { hbuf[i] = ws + off; off += (size_t)BATCH * 512; }
  float* vbuf[3];
  for (int i = 0; i < 3; ++i) { vbuf[i] = ws + off; off += (size_t)BATCH * 8; }  // 5 used, pad
  float* hpred = ws + off;               off += (size_t)BATCH * 512;
  // f16 transposed weights (half count / 2 = float slots)
  _Float16* gateWt  = (_Float16*)(ws + off); off += (size_t)DIM * (2 * DIM) / 2;      // 768 x 1536
  _Float16* VrW1t   = (_Float16*)(ws + off); off += (size_t)512 * (2 * DIM) / 2;      // 512 x 1536
  _Float16* predW1t = (_Float16*)(ws + off); off += (size_t)512 * (3 * DIM) / 2;      // 512 x 2304

  float* ecur[3] = { ebuf[0], ebuf[1], ebuf[2] };
  float* enxt[3] = { ebuf[3], ebuf[4], ebuf[5] };

  // 0. Weight prep: f16 transposed copies (once per call; reused 15x/15x/1x)
  convert_transpose_kernel<<<(DIM * 2 * DIM) / 256, 256, 0, stream>>>(gateW,  gateWt,  2 * DIM, DIM);
  convert_transpose_kernel<<<(512 * 2 * DIM) / 256, 256, 0, stream>>>(VrW1,   VrW1t,   2 * DIM, 512);
  convert_transpose_kernel<<<(512 * 3 * DIM) / 256, 256, 0, stream>>>(predW1, predW1t, 3 * DIM, 512);

  // 1. Entity max-pool (the only HBM-heavy step: ~100 MB of encoding reads)
  pool_kernel<<<BATCH * 3, 256, 0, stream>>>(enc, entpos, ent, ecur[0], ecur[1], ecur[2]);

  // 2. ent_score = sigmoid(ent @ proj_W + proj_b)
  score_kernel<<<BATCH * 3, 32, 0, stream>>>(ent, projW, projb, out + BATCH * 5);

  // 3. rel head: relu(flat(2304) @ pred_W1) @ pred_W2 + b  (flat == ent contiguous)
  //    tiles = (64/16)*(512/16) = 128 -> 32 blocks of 4 waves
  wmma_gemm<1><<<32, 128, 0, stream>>>(ent, 3 * DIM, 3 * DIM, nullptr, 0, 0,
                                       predW1t, predb1, nullptr, nullptr,
                                       hpred, 512, 512);
  small_gemm<<<(BATCH * 5 + 127) / 128, 128, 0, stream>>>(hpred, predW2, predb2, out,
                                                          BATCH, 512, 5, 0);

  // 4. Five message-passing iterations (sequential dependency; ping-pong e buffers)
  const int pa[3] = { 1, 0, 0 };   // u1: vr(e2,e3), u2: vr(e1,e3), u3: vr(e1,e2)
  const int pb[3] = { 2, 2, 1 };
  for (int it = 0; it < 5; ++it) {
    for (int p = 0; p < 3; ++p) {
      // h = relu(cat(ea, eb)(64x1536) @ Vr_W1(1536x512) + b1) : 128 tiles -> 32 blocks
      wmma_gemm<1><<<32, 128, 0, stream>>>(ecur[pa[p]], DIM, DIM, ecur[pb[p]], DIM, DIM,
                                           VrW1t, Vrb1, nullptr, nullptr,
                                           hbuf[p], 512, 512);
      // v = sigmoid(h @ Vr_W2(512x5) + b2)
      small_gemm<<<(BATCH * 5 + 127) / 128, 128, 0, stream>>>(hbuf[p], VrW2, Vrb2,
                                                              vbuf[p], BATCH, 512, 5, 2);
      // u_p = (v @ Ar_W + Ar_b) * e_p
      u_kernel<<<BATCH, 256, 0, stream>>>(vbuf[p], ArW, Arb, ecur[p], ubuf[p]);
    }
    for (int i = 0; i < 3; ++i) {
      // e_next = g*e + (1-g)*u, g = sigmoid(cat(u,e)(64x1536) @ gate_W(1536x768) + b)
      // tiles = (64/16)*(768/16) = 192 -> 48 blocks
      wmma_gemm<3><<<48, 128, 0, stream>>>(ubuf[i], DIM, DIM, ecur[i], DIM, DIM,
                                           gateWt, gateb, ecur[i], ubuf[i],
                                           enxt[i], DIM, DIM);
    }
    for (int i = 0; i < 3; ++i) { float* t = ecur[i]; ecur[i] = enxt[i]; enxt[i] = t; }
  }

  // 5. final_ent = stack([e1,e2,e3], axis=1)
  pack_final<<<(BATCH * 3 * DIM) / 256, 256, 0, stream>>>(ecur[0], ecur[1], ecur[2],
                                                          out + BATCH * 5 + BATCH * 3);
}
